// SelfAttentionBlock_283467841721
// MI455X (gfx1250) — matrix operand order repaired and checked by reading, not compiled, not run on hardware
//
#include <hip/hip_runtime.h>
#include <hip/hip_bf16.h>
#include <cmath>

// ---------------------------------------------------------------------------
// CDNA5 (gfx1250) causal self-attention, bf16 WMMA with f32 accumulate.
//   cvt:    x -> bf16 (vectorized), W -> bf16 transposed [U][D]
//   proj:   Q,K = x*W+b (bf16 [M][U]); V written transposed [B][U][T]
//   scores: S = Q K^T / sqrt(U), causal (above-diagonal tiles skipped)
//   softmax-> P bf16
//   av:     O = P V  (V^T gives direct [N][K] staging), f32 out
// All GEMMs: 128x128 tile, BK=32, 8 wave32s (4x2), wave tile 32x64,
// double-buffered LDS (integer-offset buffers so accesses stay in the DS
// address space), uint4 staging copies, ds_load_b128 fragment fetches.
// ---------------------------------------------------------------------------

typedef __attribute__((ext_vector_type(16))) __bf16 v16bf;
typedef __attribute__((ext_vector_type(8)))  float  v8f;

#define BM 128
#define BN 128
#define BK 32
#define LDT 40          // padded LDS row stride (80B: 16B-aligned, bank-conflict-free)
#define NTHREADS 256

__device__ __forceinline__ unsigned short bf16_bits(float f) {
  unsigned u = __builtin_bit_cast(unsigned, f);
  u += 0x7FFFu + ((u >> 16) & 1u);            // round to nearest even
  return (unsigned short)(u >> 16);
}
__device__ __forceinline__ __bf16 f32_to_bf16(float f) {
  unsigned short h = bf16_bits(f);
  return __builtin_bit_cast(__bf16, h);
}

union FragU { uint4 u4[2]; v16bf v; };

// Copy one 128x32 bf16 tile (row-major, ld = source leading dim) into padded
// LDS. Fully vectorized: 2x (global b128 -> ds b128) per thread.
__device__ __forceinline__ void stage_tile(const __bf16* __restrict__ src, int ld,
                                           int base, int k0, __bf16* dst, int tid) {
#pragma unroll
  for (int i = 0; i < (BM * BK) / (NTHREADS * 8); ++i) {   // 2 iters
    int idx = tid + i * NTHREADS;
    int r = idx >> 2;
    int c = (idx & 3) * 8;
    *(uint4*)&dst[r * LDT + c] =
        *(const uint4*)&src[(size_t)(base + r) * ld + (k0 + c)];
  }
}

// C(128x128) += A(rowBase..)[M x K] * Bt(colBase..)[N x K]^T, k in [0, kEnd).
// Double-buffered LDS, one barrier per K-step. The LDS is a single __shared__
// array indexed by an integer buffer offset so all accesses lower to DS ops.
__device__ __forceinline__ void wmma_mainloop(const __bf16* __restrict__ A, int lda,
                                              const __bf16* __restrict__ Bt, int ldb,
                                              int rowBase, int colBase, int kEnd,
                                              v8f acc[2][4]) {
  __shared__ __align__(16) __bf16 smem[4 * BM * LDT];
  constexpr int BUF = 2 * BM * LDT;          // one double-buffer slab (As+Bs)

  const int tid  = threadIdx.x;
  const int lane = tid & 31;
  const int wave = tid >> 5;
  const int wm = (wave >> 1) * 32;           // wave row offset in tile
  const int wn = (wave & 1) * 64;            // wave col offset in tile
  const int lr = lane & 15;
  const int lh = lane >> 4;

  stage_tile(A,  lda, rowBase, 0, smem, tid);
  stage_tile(Bt, ldb, colBase, 0, smem + BM * LDT, tid);

  int curOff = 0;
  for (int k0 = 0; k0 < kEnd; k0 += BK) {
    __syncthreads();
    const int nxtOff = BUF - curOff;
    if (k0 + BK < kEnd) {                    // prefetch next K-slab
      stage_tile(A,  lda, rowBase, k0 + BK, smem + nxtOff, tid);
      stage_tile(Bt, ldb, colBase, k0 + BK, smem + nxtOff + BM * LDT, tid);
    }
    const __bf16* As = smem + curOff;
    const __bf16* Bs = smem + curOff + BM * LDT;

    FragU af[2], bfg[4];
#pragma unroll
    for (int fm = 0; fm < 2; ++fm) {         // A frag: K = {8*lh..+7, 8*lh+16..+23}
      const __bf16* p = &As[(wm + fm * 16 + lr) * LDT + lh * 8];
      af[fm].u4[0] = *(const uint4*)p;
      af[fm].u4[1] = *(const uint4*)(p + 16);
    }
#pragma unroll
    for (int fn = 0; fn < 4; ++fn) {         // B frag: K = {8*lh..+7, 8*lh+16..+23}
      const __bf16* p = &Bs[(wn + fn * 16 + lr) * LDT + lh * 8];
      bfg[fn].u4[0] = *(const uint4*)p;
      bfg[fn].u4[1] = *(const uint4*)(p + 16);
    }
#pragma unroll
    for (int fm = 0; fm < 2; ++fm)
#pragma unroll
      for (int fn = 0; fn < 4; ++fn)
        acc[fm][fn] = __builtin_amdgcn_wmma_f32_16x16x32_bf16(
            false, af[fm].v, false, bfg[fn].v, (short)0, acc[fm][fn], false, false);
    curOff = nxtOff;
  }
}

// ---------------------------------------------------------------------------
// f32 -> bf16 bulk convert, 4 elems/thread (b128 load, b64 store)
__global__ void __launch_bounds__(256) cvt4_kernel(const float* __restrict__ in,
                                                   __bf16* __restrict__ out, int n4) {
  int i = blockIdx.x * 256 + threadIdx.x;
  if (i >= n4) return;
  float4 f = *(const float4*)(in + (size_t)i * 4);
  uint2 pk;
  pk.x = (unsigned)bf16_bits(f.x) | ((unsigned)bf16_bits(f.y) << 16);
  pk.y = (unsigned)bf16_bits(f.z) | ((unsigned)bf16_bits(f.w) << 16);
  *(uint2*)(out + (size_t)i * 4) = pk;
}

// f32 [R][C] -> bf16 [C][R] (transposing convert, 32x32 LDS tiles)
__global__ void __launch_bounds__(256) cvtT_kernel(const float* __restrict__ in,
                                                   __bf16* __restrict__ out,
                                                   int R, int C) {
  __shared__ float tile[32][33];
  const int c0 = blockIdx.x * 32, r0 = blockIdx.y * 32;
  const int tx = threadIdx.x, ty = threadIdx.y;
  for (int i = ty; i < 32; i += 8)
    tile[i][tx] = in[(size_t)(r0 + i) * C + (c0 + tx)];
  __syncthreads();
  for (int i = ty; i < 32; i += 8)
    out[(size_t)(c0 + i) * R + (r0 + tx)] = f32_to_bf16(tile[tx][i]);
}

// C = A[M x K] * Wt[N x K]^T + bias.
// VTRANS=false: out bf16 [M][N].  VTRANS=true: out bf16 [B][N][T] (V^T), with
// the 8 accumulator rows per lane packed into one 16B store.
template <bool VTRANS>
__global__ void __launch_bounds__(256) proj_kernel(const __bf16* __restrict__ X,
                                                   const __bf16* __restrict__ Wt,
                                                   const float* __restrict__ bias,
                                                   __bf16* __restrict__ out,
                                                   int M, int N, int K, int T) {
  const int rowBase = blockIdx.y * BM, colBase = blockIdx.x * BN;
  v8f acc[2][4] = {};
  wmma_mainloop(X, K, Wt, K, rowBase, colBase, K, acc);

  const int lane = threadIdx.x & 31, wave = threadIdx.x >> 5;
  const int wm = (wave >> 1) * 32, wn = (wave & 1) * 64;
  const int lr = lane & 15, lh = lane >> 4;
#pragma unroll
  for (int fm = 0; fm < 2; ++fm)
#pragma unroll
    for (int fn = 0; fn < 4; ++fn) {
      const int col = colBase + wn + fn * 16 + lr;
      const float b = bias[col];
      if (VTRANS) {
        const int rowStart = rowBase + wm + fm * 16 + lh * 8;   // 8 consecutive rows
        const int bb = rowStart / T, tt = rowStart % T;
        uint4 pk;
        pk.x = (unsigned)bf16_bits(acc[fm][fn][0] + b) | ((unsigned)bf16_bits(acc[fm][fn][1] + b) << 16);
        pk.y = (unsigned)bf16_bits(acc[fm][fn][2] + b) | ((unsigned)bf16_bits(acc[fm][fn][3] + b) << 16);
        pk.z = (unsigned)bf16_bits(acc[fm][fn][4] + b) | ((unsigned)bf16_bits(acc[fm][fn][5] + b) << 16);
        pk.w = (unsigned)bf16_bits(acc[fm][fn][6] + b) | ((unsigned)bf16_bits(acc[fm][fn][7] + b) << 16);
        *(uint4*)&out[((size_t)bb * N + col) * T + tt] = pk;
      } else {
#pragma unroll
        for (int e = 0; e < 8; ++e) {
          const int row = rowBase + wm + fm * 16 + lh * 8 + e;
          out[(size_t)row * N + col] = f32_to_bf16(acc[fm][fn][e] + b);
        }
      }
    }
}

// S[f32, TxT] = scale * Q K^T, causal-masked; above-diagonal tiles skipped.
__global__ void __launch_bounds__(256) scores_kernel(const __bf16* __restrict__ Q,
                                                     const __bf16* __restrict__ Km,
                                                     float* __restrict__ S,
                                                     int T, int U, float scale) {
  const int b = blockIdx.z;
  Q  += (size_t)b * T * U;
  Km += (size_t)b * T * U;
  S  += (size_t)b * T * T;
  const int rowBase = blockIdx.y * BM, colBase = blockIdx.x * BN;

  if (colBase >= rowBase + BM) {             // tile entirely above diagonal
    const float4 neg = { -1e9f, -1e9f, -1e9f, -1e9f };
    for (int i = threadIdx.x; i < (BM * BN) / 4; i += NTHREADS) {
      int r = i >> 5, c = (i & 31) * 4;
      *(float4*)&S[(size_t)(rowBase + r) * T + (colBase + c)] = neg;
    }
    return;
  }

  v8f acc[2][4] = {};
  wmma_mainloop(Q, U, Km, U, rowBase, colBase, U, acc);   // B = K, [N][K] direct

  const bool diag = (colBase == rowBase);
  const int lane = threadIdx.x & 31, wave = threadIdx.x >> 5;
  const int wm = (wave >> 1) * 32, wn = (wave & 1) * 64;
  const int lr = lane & 15, lh = lane >> 4;
#pragma unroll
  for (int fm = 0; fm < 2; ++fm)
#pragma unroll
    for (int fn = 0; fn < 4; ++fn) {
      const int col = colBase + wn + fn * 16 + lr;
#pragma unroll
      for (int e = 0; e < 8; ++e) {
        const int row = rowBase + wm + fm * 16 + lh * 8 + e;
        float v = acc[fm][fn][e] * scale;
        if (diag && col > row) v = -1e9f;
        S[(size_t)row * T + col] = v;
      }
    }
}

// Row softmax: one block per (b,t) row; P stored bf16.
__global__ void __launch_bounds__(256) softmax_kernel(const float* __restrict__ S,
                                                      __bf16* __restrict__ P, int T) {
  const size_t row = blockIdx.x;
  S += row * T;
  P += row * T;
  const int tid = threadIdx.x;
  const int nv = T / 256;                    // T=512 -> 2 elems/thread
  float v[4];
  float m = -3.4e38f;
  for (int j = 0; j < nv; ++j) { v[j] = S[tid + j * 256]; m = fmaxf(m, v[j]); }

  __shared__ float red[256];
  red[tid] = m; __syncthreads();
  for (int s = 128; s > 0; s >>= 1) {
    if (tid < s) red[tid] = fmaxf(red[tid], red[tid + s]);
    __syncthreads();
  }
  m = red[0]; __syncthreads();

  float sum = 0.f;
  for (int j = 0; j < nv; ++j) { v[j] = __expf(v[j] - m); sum += v[j]; }
  red[tid] = sum; __syncthreads();
  for (int s = 128; s > 0; s >>= 1) {
    if (tid < s) red[tid] += red[tid + s];
    __syncthreads();
  }
  const float inv = 1.f / red[0];
  for (int j = 0; j < nv; ++j) P[tid + j * 256] = f32_to_bf16(v[j] * inv);
}

// O[f32, TxU] = P[bf16, TxT] * V; V held transposed [U][T] => direct staging.
// K-loop truncated by causality.
__global__ void __launch_bounds__(256) av_kernel(const __bf16* __restrict__ P,
                                                 const __bf16* __restrict__ Vt,
                                                 float* __restrict__ O,
                                                 int T, int U) {
  const int b = blockIdx.z;
  P  += (size_t)b * T * T;
  Vt += (size_t)b * U * T;
  O  += (size_t)b * T * U;
  const int rowBase = blockIdx.y * BM, colBase = blockIdx.x * BN;
  int kEnd = rowBase + BM;                   // causal truncation
  if (kEnd > T) kEnd = T;

  v8f acc[2][4] = {};
  wmma_mainloop(P, T, Vt, T, rowBase, colBase, kEnd, acc);

  const int lane = threadIdx.x & 31, wave = threadIdx.x >> 5;
  const int wm = (wave >> 1) * 32, wn = (wave & 1) * 64;
  const int lr = lane & 15, lh = lane >> 4;
#pragma unroll
  for (int fm = 0; fm < 2; ++fm)
#pragma unroll
    for (int fn = 0; fn < 4; ++fn) {
      const int col = colBase + wn + fn * 16 + lr;
#pragma unroll
      for (int e = 0; e < 8; ++e) {
        const int row = rowBase + wm + fm * 16 + lh * 8 + e;
        O[(size_t)row * U + col] = acc[fm][fn][e];
      }
    }
}

// ---------------------------------------------------------------------------
extern "C" void kernel_launch(void* const* d_in, const int* in_sizes, int n_in,
                              void* d_out, int out_size, void* d_ws, size_t ws_size,
                              hipStream_t stream) {
  (void)in_sizes; (void)n_in; (void)out_size; (void)ws_size;
  const int B = 32, T = 512, D = 256, U = 512;
  const int M = B * T;

  const float* x  = (const float*)d_in[0];
  const float* Wq = (const float*)d_in[1];
  const float* bq = (const float*)d_in[2];
  const float* Wk = (const float*)d_in[3];
  const float* bk = (const float*)d_in[4];
  const float* Wv = (const float*)d_in[5];
  const float* bv = (const float*)d_in[6];
  float* out = (float*)d_out;

  char* ws = (char*)d_ws;
  size_t off = 0;
  auto carve = [&](size_t bytes) {
    char* p = ws + off;
    off += (bytes + 255) & ~(size_t)255;
    return p;
  };
  __bf16* xbf = (__bf16*)carve((size_t)M * D * 2);
  __bf16* Wqt = (__bf16*)carve((size_t)D * U * 2);   // [U][D]
  __bf16* Wkt = (__bf16*)carve((size_t)D * U * 2);
  __bf16* Wvt = (__bf16*)carve((size_t)D * U * 2);
  __bf16* Qb  = (__bf16*)carve((size_t)M * U * 2);   // [M][U]
  __bf16* Kb  = (__bf16*)carve((size_t)M * U * 2);   // [M][U]
  __bf16* Vtb = (__bf16*)carve((size_t)M * U * 2);   // [B][U][T]
  float*  Sb  = (float*) carve((size_t)B * T * T * 4);
  __bf16* Pb  = (__bf16*)carve((size_t)B * T * T * 2);

  // 1) converts: x -> bf16; W -> bf16 transposed [U][D]
  {
    int n4 = (M * D) / 4;
    cvt4_kernel<<<dim3((n4 + 255) / 256), 256, 0, stream>>>(x, xbf, n4);
    dim3 tg(U / 32, D / 32);
    dim3 tb(32, 8);
    cvtT_kernel<<<tg, tb, 0, stream>>>(Wq, Wqt, D, U);
    cvtT_kernel<<<tg, tb, 0, stream>>>(Wk, Wkt, D, U);
    cvtT_kernel<<<tg, tb, 0, stream>>>(Wv, Wvt, D, U);
  }

  // 2) projections
  {
    dim3 grid(U / BN, M / BM);
    proj_kernel<false><<<grid, NTHREADS, 0, stream>>>(xbf, Wqt, bq, Qb,  M, U, D, T);
    proj_kernel<false><<<grid, NTHREADS, 0, stream>>>(xbf, Wkt, bk, Kb,  M, U, D, T);
    proj_kernel<true ><<<grid, NTHREADS, 0, stream>>>(xbf, Wvt, bv, Vtb, M, U, D, T);
  }

  // 3) causal scaled scores
  {
    dim3 grid(T / BN, T / BM, B);
    const float scale = 1.0f / sqrtf((float)U);
    scores_kernel<<<grid, NTHREADS, 0, stream>>>(Qb, Kb, Sb, T, U, scale);
  }

  // 4) softmax -> bf16 probs
  softmax_kernel<<<dim3(B * T), 256, 0, stream>>>(Sb, Pb, T);

  // 5) O = P V
  {
    dim3 grid(U / BN, T / BM, B);
    av_kernel<<<grid, NTHREADS, 0, stream>>>(Pb, Vtb, out, T, U);
  }
}
